// MultiHeadTCDCSelfSGAttention_66683662237914
// MI455X (gfx1250) — matrix operand order repaired and checked by reading, not compiled, not run on hardware
//
#include <hip/hip_runtime.h>
#include <hip/hip_bf16.h>

typedef __attribute__((ext_vector_type(16))) _Float16 v16h;
typedef __attribute__((ext_vector_type(8)))  float    v8f;

#define B_     4
#define DEPTH_ 160
#define FS_    4
#define C_     96
#define NH_    4
#define N_     (DEPTH_*FS_*FS_)   // 2560
#define HD_    (C_/NH_)           // 24
#define HDP_   32                 // head dim padded to WMMA K
#define THETA_ 0.6f
#define BN_EPS_ 1e-5f
#define KCONV_ 2688               // 27 taps * 96 cin + 96 diff = 84 chunks of 32
#define NKC_   (KCONV_/32)        // 84 K-chunks

union AV { v16h v; unsigned u[8]; _Float16 h[16]; };

// 16-lane-group xor butterfly via DPP16 row_xmask (pure VALU, no LDS traffic)
template<int CTRL>
__device__ __forceinline__ float dppx(float x) {
    return __int_as_float(__builtin_amdgcn_update_dpp(
        0, __float_as_int(x), CTRL, 0xf, 0xf, true));
}
__device__ __forceinline__ float rowmax16(float x) {
    x = fmaxf(x, dppx<0x161>(x));   // row_xmask:1
    x = fmaxf(x, dppx<0x162>(x));   // row_xmask:2
    x = fmaxf(x, dppx<0x164>(x));   // row_xmask:4
    x = fmaxf(x, dppx<0x168>(x));   // row_xmask:8
    return x;
}
__device__ __forceinline__ float rowsum16(float x) {
    x += dppx<0x161>(x);
    x += dppx<0x162>(x);
    x += dppx<0x164>(x);
    x += dppx<0x168>(x);
    return x;
}

// ---------------------------------------------------------------------------
// Kernel 0: pack weights for the WMMA conv-GEMM.
//  wqt/wkt: f16 [cout][2688], k = tap*96+cin for taps 0..26, then 96 entries
//           of -theta * (sum of kd=0 and kd=2 taps)  (TCDC diff folded into K)
//  wvt/woh: f16 row-major 96x96
// ---------------------------------------------------------------------------
__global__ void prep_kernel(const float* __restrict__ wq, const float* __restrict__ wk,
                            const float* __restrict__ wv, const float* __restrict__ wo,
                            _Float16* __restrict__ wqt, _Float16* __restrict__ wkt,
                            _Float16* __restrict__ wvt, _Float16* __restrict__ woh) {
    int i = blockIdx.x * blockDim.x + threadIdx.x;
    if (i < C_*KCONV_) {
        int cout = i / KCONV_;
        int k    = i - cout*KCONV_;
        float vq, vk;
        if (k < 27*C_) {
            int tap = k / C_;
            int cin = k - tap*C_;
            vq = wq[((size_t)cout*C_ + cin)*27 + tap];
            vk = wk[((size_t)cout*C_ + cin)*27 + tap];
        } else {
            int cin = k - 27*C_;
            const float* pq = wq + ((size_t)cout*C_ + cin)*27;
            const float* pk = wk + ((size_t)cout*C_ + cin)*27;
            float sq = 0.f, sk = 0.f;
#pragma unroll
            for (int t = 0; t < 9; ++t) { sq += pq[t] + pq[18+t]; sk += pk[t] + pk[18+t]; }
            vq = -THETA_*sq;
            vk = -THETA_*sk;
        }
        wqt[(size_t)cout*KCONV_ + k] = (_Float16)vq;
        wkt[(size_t)cout*KCONV_ + k] = (_Float16)vk;
    }
    if (i < C_*C_) {
        wvt[i] = (_Float16)wv[i];
        woh[i] = (_Float16)wo[i];
    }
}

// ---------------------------------------------------------------------------
// Kernel 1: q/k/v via WMMA im2col GEMM. One block per (depth slice, batch),
// 6 waves = 6 cout tiles of 16. x staged as f16 in LDS; K loop of 84 chunks:
// A tile (16 tokens x 32 K) from LDS, B rows from packed weights; q & k share
// every A, v rides on the 3 center-tap chunks. BN folded as scale/shift on C.
// ---------------------------------------------------------------------------
__global__ void qkv_kernel(const float* __restrict__ x,
    const _Float16* __restrict__ wqt, const _Float16* __restrict__ wkt,
    const _Float16* __restrict__ wvt,
    const float* __restrict__ qg, const float* __restrict__ qb,
    const float* __restrict__ qm, const float* __restrict__ qvv,
    const float* __restrict__ kg, const float* __restrict__ kb,
    const float* __restrict__ km, const float* __restrict__ kvv,
    _Float16* __restrict__ qh, _Float16* __restrict__ kh, _Float16* __restrict__ vt)
{
    const int d = blockIdx.x;
    const int b = blockIdx.y;
    __shared__ __align__(16) _Float16 xs[3][16][C_];   // 9 KB, f16-staged input

    for (int e = threadIdx.x; e < 3*16*C_; e += blockDim.x) {
        int dz  = e / (16*C_);
        int rem = e - dz*16*C_;
        int tok = rem / C_;
        int c   = rem - tok*C_;
        int dd  = d + dz - 1;
        float val = (dd >= 0 && dd < DEPTH_)
                  ? x[((size_t)b*N_ + (size_t)dd*16 + tok)*C_ + c] : 0.f;
        xs[dz][tok][c] = (_Float16)val;
    }
    __syncthreads();

    const int lane     = threadIdx.x & 31;
    const int coutTile = threadIdx.x >> 5;        // wave id 0..5
    const int l16      = lane & 15;
    const int kb4      = (lane >> 4) * 4;
    const int h0 = l16 >> 2, w0 = l16 & 3;        // token row this lane feeds to A

    // B-operand row bases: col = cout = coutTile*16 + l16, K pairs contiguous
    const int cout = coutTile*16 + l16;
    const _Float16* bqBase = wqt + (size_t)cout*KCONV_ + ((lane >> 4) << 4);
    const _Float16* bkBase = wkt + (size_t)cout*KCONV_ + ((lane >> 4) << 4);
    const _Float16* bvBase = wvt + (size_t)cout*C_     + ((lane >> 4) << 4);

    v8f accq, acck, accv;
#pragma unroll
    for (int r = 0; r < 8; ++r) { accq[r] = 0.f; acck[r] = 0.f; accv[r] = 0.f; }

    for (int kc = 0; kc < NKC_; ++kc) {
        // chunk -> (tap, cin0); chunks 81..83 are the folded diff (center tap)
        int kd, khh, kww, cin0;
        if (kc < 81) {
            int tap = kc / 3;
            cin0    = (kc - tap*3) * 32;
            kd      = tap / 9;
            int r2  = tap - kd*9;
            khh     = r2 / 3;
            kww     = r2 - khh*3;
        } else {
            kd = 1; khh = 1; kww = 1; cin0 = (kc - 81) * 32;
        }
        int hh = h0 + khh - 1, ww = w0 + kww - 1;
        bool valid = (hh >= 0) & (hh < 4) & (ww >= 0) & (ww < 4);

        AV a;
        if (valid) {
            const unsigned* ap = (const unsigned*)&xs[kd][hh*4 + ww][cin0];
#pragma unroll
            for (int j = 0; j < 4; ++j) { a.u[j] = ap[2*kb4 + j]; a.u[4+j] = ap[2*kb4 + 4 + j]; }
        } else {
#pragma unroll
            for (int j = 0; j < 8; ++j) a.u[j] = 0u;
        }

        AV bq, bk;
        {
            const unsigned* pq = (const unsigned*)(bqBase + (size_t)kc*32);
            const unsigned* pk = (const unsigned*)(bkBase + (size_t)kc*32);
#pragma unroll
            for (int j = 0; j < 8; ++j) { bq.u[j] = pq[j]; bk.u[j] = pk[j]; }
        }
        accq = __builtin_amdgcn_wmma_f32_16x16x32_f16(false, a.v, false, bq.v, (short)0, accq, false, false);
        acck = __builtin_amdgcn_wmma_f32_16x16x32_f16(false, a.v, false, bk.v, (short)0, acck, false, false);

        if (kc >= 39 && kc < 42) {   // center tap: 1x1x1 v conv shares this A
            AV bv;
            const unsigned* pv = (const unsigned*)(bvBase + (size_t)(kc - 39)*32);
#pragma unroll
            for (int j = 0; j < 8; ++j) bv.u[j] = pv[j];
            accv = __builtin_amdgcn_wmma_f32_16x16x32_f16(false, a.v, false, bv.v, (short)0, accv, false, false);
        }
    }

    // eval-BN as scale/shift on the f32 C tile; store f16 q/k (padded) and v^T
    float qsc = qg[cout] * rsqrtf(qvv[cout] + BN_EPS_);
    float ksc = kg[cout] * rsqrtf(kvv[cout] + BN_EPS_);
    float qsh = qb[cout] - qm[cout]*qsc;
    float ksh = kb[cout] - km[cout]*ksc;
    int head = cout / HD_;
    int dim  = cout - head*HD_;
    size_t bh = (size_t)b*NH_ + head;
#pragma unroll
    for (int r = 0; r < 8; ++r) {
        int tok = r + ((lane >> 4) << 3);       // C layout: M = r + 8*(lane/16)
        int n   = d*16 + tok;
        qh[(bh*N_ + n)*HDP_ + dim] = (_Float16)(accq[r]*qsc + qsh);
        kh[(bh*N_ + n)*HDP_ + dim] = (_Float16)(acck[r]*ksc + ksh);
        vt[(bh*HD_ + dim)*N_ + n]  = (_Float16)accv[r];
    }

    // zero the head-dim padding (dims 24..31) of q/k for this slice
    for (int p = threadIdx.x; p < 16*NH_*(HDP_-HD_); p += blockDim.x) {
        int tok  = p >> 5;
        int rr   = p & 31;
        int hd2  = rr >> 3;
        int dim2 = HD_ + (rr & 7);
        int n    = d*16 + tok;
        size_t bh2 = (size_t)b*NH_ + hd2;
        qh[(bh2*N_ + n)*HDP_ + dim2] = (_Float16)0.f;
        kh[(bh2*N_ + n)*HDP_ + dim2] = (_Float16)0.f;
    }
}

// ---------------------------------------------------------------------------
// Kernel 2: flash attention. One wave32 per (b, h, 16-query tile), 4 waves/blk.
// Row reductions via DPP16 row_xmask (VALU); P routed C->A layout via 512B LDS.
// ---------------------------------------------------------------------------
__global__ void attn_kernel(const _Float16* __restrict__ qh, const _Float16* __restrict__ kh,
                            const _Float16* __restrict__ vt, const float* __restrict__ sg,
                            _Float16* __restrict__ ao)
{
    __shared__ __align__(16) _Float16 pls[4][16][16];
    const int lane  = threadIdx.x & 31;
    const int wslot = threadIdx.x >> 5;
    int wid  = blockIdx.x*4 + wslot;
    int qt   = wid % (N_/16);
    int rest = wid / (N_/16);
    int h    = rest & 3;
    int b    = rest >> 2;
    const float inv = 1.0f / sg[0];
    size_t bh = (size_t)b*NH_ + h;

    AV qa;
    {
        const unsigned* q32 = (const unsigned*)(qh + (bh*N_ + (size_t)qt*16 + (lane & 15))*HDP_);
        int kb4 = (lane >> 4) * 4;
#pragma unroll
        for (int j = 0; j < 4; ++j) { qa.u[j] = q32[2*kb4 + j]; qa.u[4+j] = q32[2*kb4 + 4 + j]; }
    }

    float mrow[8], lrow[8];
    v8f o0, o1;
#pragma unroll
    for (int r = 0; r < 8; ++r) { mrow[r] = -1e30f; lrow[r] = 0.f; o0[r] = 0.f; o1[r] = 0.f; }

    for (int kt = 0; kt < N_/16; ++kt) {
        if (kt + 1 < N_/16)
            __builtin_prefetch(kh + (bh*N_ + (size_t)(kt+1)*16)*HDP_, 0, 3);

        AV kbm;
        {
            const unsigned* k32 = (const unsigned*)(kh + (bh*N_ + (size_t)kt*16 + (lane & 15))*HDP_);
            int kb8 = (lane >> 4) * 8;
#pragma unroll
            for (int j = 0; j < 8; ++j) kbm.u[j] = k32[kb8 + j];
        }
        v8f zc;
#pragma unroll
        for (int r = 0; r < 8; ++r) zc[r] = 0.f;
        v8f s = __builtin_amdgcn_wmma_f32_16x16x32_f16(false, qa.v, false, kbm.v,
                                                       (short)0, zc, false, false);

#pragma unroll
        for (int r = 0; r < 8; ++r) {
            float sv   = s[r] * inv;
            float rm   = rowmax16(sv);
            float mn   = fmaxf(mrow[r], rm);
            float corr = __expf(mrow[r] - mn);
            float p    = __expf(sv - mn);
            float rs   = rowsum16(p);
            lrow[r] = lrow[r]*corr + rs;
            o0[r]  *= corr;
            o1[r]  *= corr;
            mrow[r] = mn;
            pls[wslot][r + ((lane >> 4) << 3)][lane & 15] = (_Float16)p;
        }
        asm volatile("s_wait_dscnt 0" ::: "memory");

        AV pa;
        {
            const unsigned* p32 = (const unsigned*)&pls[wslot][lane & 15][0];
            int kb4 = (lane >> 4) * 4;
#pragma unroll
            for (int j = 0; j < 4; ++j) { pa.u[j] = kb4 ? 0u : p32[j]; pa.u[4+j] = kb4 ? 0u : p32[4 + j]; }
        }

        AV vb0, vb1;
#pragma unroll
        for (int j = 0; j < 8; ++j) { vb0.u[j] = 0u; vb1.u[j] = 0u; }
        if (lane < 16) {
            const unsigned* vp0 = (const unsigned*)(vt + (bh*HD_ + lane)*N_ + (size_t)kt*16);
#pragma unroll
            for (int j = 0; j < 8; ++j) vb0.u[j] = vp0[j];
            int d1 = 16 + lane;
            if (d1 < HD_) {
                const unsigned* vp1 = (const unsigned*)(vt + (bh*HD_ + d1)*N_ + (size_t)kt*16);
#pragma unroll
                for (int j = 0; j < 8; ++j) vb1.u[j] = vp1[j];
            }
        }
        o0 = __builtin_amdgcn_wmma_f32_16x16x32_f16(false, pa.v, false, vb0.v, (short)0, o0, false, false);
        o1 = __builtin_amdgcn_wmma_f32_16x16x32_f16(false, pa.v, false, vb1.v, (short)0, o1, false, false);
    }

#pragma unroll
    for (int r = 0; r < 8; ++r) {
        float invl = 1.0f / lrow[r];
        int token  = qt*16 + r + ((lane >> 4) << 3);
        size_t row = ((size_t)b*N_ + token)*C_ + (size_t)h*HD_;
        ao[row + (lane & 15)] = (_Float16)(o0[r] * invl);
        int d1 = 16 + (lane & 15);
        if (d1 < HD_) ao[row + d1] = (_Float16)(o1[r] * invl);
    }
}

// ---------------------------------------------------------------------------
// Kernel 3: out projection (10240 x 96 x 96) + bias, WMMA tiles, f32 output
// ---------------------------------------------------------------------------
__global__ void proj_kernel(const _Float16* __restrict__ ao, const _Float16* __restrict__ woh,
                            const float* __restrict__ bo, float* __restrict__ out)
{
    const int lane = threadIdx.x & 31;
    int wid = blockIdx.x*4 + (threadIdx.x >> 5);
    const int MT = (B_*N_)/16;
    int mt = wid % MT;
    int nt = wid / MT;

    v8f acc;
#pragma unroll
    for (int r = 0; r < 8; ++r) acc[r] = 0.f;

    for (int ktile = 0; ktile < 3; ++ktile) {
        AV a, bm;
        const unsigned* a32 =
            (const unsigned*)(ao + (size_t)(mt*16 + (lane & 15))*C_ + ktile*32);
        int kb4 = (lane >> 4) * 4;
#pragma unroll
        for (int j = 0; j < 4; ++j) { a.u[j] = a32[2*kb4 + j]; a.u[4+j] = a32[2*kb4 + 4 + j]; }

        const unsigned* b32 =
            (const unsigned*)(woh + (size_t)(nt*16 + (lane & 15))*C_ + ktile*32 + ((lane >> 4) << 4));
#pragma unroll
        for (int j = 0; j < 8; ++j) bm.u[j] = b32[j];

        acc = __builtin_amdgcn_wmma_f32_16x16x32_f16(false, a.v, false, bm.v, (short)0, acc, false, false);
    }

#pragma unroll
    for (int r = 0; r < 8; ++r) {
        int row = mt*16 + r + ((lane >> 4) << 3);
        int col = nt*16 + (lane & 15);
        out[(size_t)row*C_ + col] = acc[r] + bo[col];
    }
}

// ---------------------------------------------------------------------------
extern "C" void kernel_launch(void* const* d_in, const int* in_sizes, int n_in,
                              void* d_out, int out_size, void* d_ws, size_t ws_size,
                              hipStream_t stream) {
    const float* x   = (const float*)d_in[0];
    const float* sg  = (const float*)d_in[1];
    const float* wq  = (const float*)d_in[2];
    const float* qg  = (const float*)d_in[3];
    const float* qb  = (const float*)d_in[4];
    const float* qm  = (const float*)d_in[5];
    const float* qvv = (const float*)d_in[6];
    const float* wk  = (const float*)d_in[7];
    const float* kg  = (const float*)d_in[8];
    const float* kb  = (const float*)d_in[9];
    const float* km  = (const float*)d_in[10];
    const float* kvv = (const float*)d_in[11];
    const float* wv  = (const float*)d_in[12];
    const float* wo  = (const float*)d_in[13];
    const float* bo  = (const float*)d_in[14];

    char* ws = (char*)d_ws;
    size_t off = 0;
    auto take = [&](size_t bytes) { char* p = ws + off; off += (bytes + 255) & ~(size_t)255; return p; };
    _Float16* wqt = (_Float16*)take((size_t)C_*KCONV_*2);
    _Float16* wkt = (_Float16*)take((size_t)C_*KCONV_*2);
    _Float16* wvt = (_Float16*)take((size_t)C_*C_*2);
    _Float16* woh = (_Float16*)take((size_t)C_*C_*2);
    _Float16* qh  = (_Float16*)take((size_t)B_*NH_*N_*HDP_*2);
    _Float16* kh  = (_Float16*)take((size_t)B_*NH_*N_*HDP_*2);
    _Float16* vt  = (_Float16*)take((size_t)B_*NH_*HD_*N_*2);
    _Float16* ao  = (_Float16*)take((size_t)B_*N_*C_*2);
    (void)ws_size; (void)n_in; (void)in_sizes; (void)out_size;

    prep_kernel<<<(C_*KCONV_ + 255)/256, 256, 0, stream>>>(wq, wk, wv, wo, wqt, wkt, wvt, woh);

    qkv_kernel<<<dim3(DEPTH_, B_), 192, 0, stream>>>(
        x, wqt, wkt, wvt,
        qg, qb, qm, qvv, kg, kb, km, kvv,
        qh, kh, vt);

    attn_kernel<<<(B_*NH_*(N_/16))/4, 128, 0, stream>>>(qh, kh, vt, sg, ao);

    proj_kernel<<<((B_*N_/16)*6)/4, 128, 0, stream>>>(ao, woh, bo, (float*)d_out);
}
